// LinearLUT_50895362458169
// MI455X (gfx1250) — matrix operand order repaired and checked by reading, not compiled, not run on hardware
//
#include <hip/hip_runtime.h>
#include <hip/hip_bf16.h>

typedef __attribute__((ext_vector_type(16))) _Float16 v16h;
typedef __attribute__((ext_vector_type(8)))  float    v8f;

#define LEVELS 2
#define KBITS  4
#define KK     16
#define IN     128
#define OUT    128
#define BATCH  128
#define T      (IN*OUT)

// ---------------- Kernel A: residual-sign bit packing ----------------
// spack layout: unsigned [LEVELS][BATCH][4]  (128 bits per (l,b))
__global__ void signs_kernel(const float* __restrict__ x,
                             const float* __restrict__ means,
                             unsigned* __restrict__ spack) {
    __shared__ unsigned char sb[2][IN];
    const int b = blockIdx.x;
    const int i = threadIdx.x;
    const float m0 = fabsf(means[0]);
    float v = x[b*IN + i];
    int s0 = (v >= 0.0f) ? 1 : 0;
    float r = v - (s0 ? m0 : -m0);
    int s1 = (r >= 0.0f) ? 1 : 0;
    sb[0][i] = (unsigned char)s0;
    sb[1][i] = (unsigned char)s1;
    __syncthreads();
    if (i < 8) {
        int l = i >> 2, w = i & 3;
        unsigned bits = 0;
        for (int j = 0; j < 32; ++j) bits |= ((unsigned)sb[l][w*32 + j]) << j;
        spack[(l*BATCH + b)*4 + w] = bits;
    }
}

// ---------------- Kernel B: WMMA weight transform  G_l = W * M_l^T ----
// g layout: float [LEVELS][T][KK]
// M_l[p][c] = a^(4-h) b^h,  h = popcount(p^c),  a=0.5(1+m), b=0.5(1-m)
//           = a^4 * (b/a)^h   -> branchless via 3 selected multiplies.
__global__ void wmma_transform_kernel(const float* __restrict__ weight,
                                      const float* __restrict__ means,
                                      float* __restrict__ g) {
    const int tile  = blockIdx.x;      // T/16 tiles of 16 weight rows
    const int lane  = threadIdx.x;     // wave32, EXEC all-1s
    const int row16 = lane & 15;
    const int hi    = lane >> 4;
    const int row   = tile*16 + row16;
    const int cbase = hi * 8;          // A layout: lo lanes K0..15, hi lanes zero (K 16..31)

    // A matrix 16x32 f16: lo lanes elements 0..15 = real K 0..15, hi lanes = zero padding (K 16..31)
    v16h a;
#pragma unroll
    for (int e = 0; e < 16; ++e) a[e] = (_Float16)0.0f;
#pragma unroll
    for (int e = 0; e < 16; ++e) a[e] = hi ? (_Float16)0.0f : (_Float16)weight[row*KK + e];

#pragma unroll
    for (int l = 0; l < LEVELS; ++l) {
        float m  = fabsf(means[l]);
        float pa = 0.5f*(1.0f + m);
        float pb = 0.5f*(1.0f - m);
        float r1 = pb / pa;            // ratio < 1
        float r2 = r1 * r1;
        float r4 = r2 * r2;
        float pw0 = pa*pa*pa*pa;
        // Fold the K>=16 zero padding (hi lanes) into the base factor once.
        float base = (hi == 0) ? pw0 : 0.0f;

        // B matrix 32x16 f16: lane = column p (lanes 0..15), element e = K = c.
        v16h bm;
#pragma unroll
        for (int c = 0; c < 16; ++c) {
            int h = __popc((row16 ^ c) & 15);     // 0..4
            float val = base;
            val *= (h & 1) ? r1 : 1.0f;
            val *= (h & 2) ? r2 : 1.0f;
            val *= (h & 4) ? r4 : 1.0f;
            bm[c] = (_Float16)val;
        }
        v8f c8 = {};
        v8f d = __builtin_amdgcn_wmma_f32_16x16x32_f16(
            /*neg_a=*/false, a, /*neg_b=*/false, bm,
            /*c_mod=*/(short)0, c8, /*reuse_a=*/false, /*reuse_b=*/false);

        float* gl = g + (size_t)l * T * KK;
#pragma unroll
        for (int rr = 0; rr < 8; ++rr) {
            int M = rr + (hi ? 8 : 0);
            gl[(size_t)(tile*16 + M)*KK + row16] = d[rr];
        }
    }
}

// ---------------- Kernel C: LUT gather + reduce over i ----------------
__global__ void lut_gather_kernel(const float* __restrict__ g,
                                  const unsigned* __restrict__ spack,
                                  const int* __restrict__ input_mask,
                                  const float* __restrict__ bias,
                                  float* __restrict__ out) {
    __shared__ float    gs[LEVELS][IN][KK];   // 16 KB: LUTs for this output o
    __shared__ unsigned mks[IN];              // 4 packed mask bytes per i
    __shared__ float    partial[256];

    const int o   = blockIdx.x;
    const int tid = threadIdx.x;

    // Load g tiles: contiguous 2048 floats per level for this o
    {
        const float4* gsrc = (const float4*)g;
        float4* gdst = (float4*)&gs[0][0][0];
        for (int j = tid; j < LEVELS*512; j += 256) {
            int l  = j >> 9;
            int jj = j & 511;
            gdst[l*512 + jj] = gsrc[(size_t)l*(T*KK/4) + (size_t)o*512 + jj];
        }
    }
    if (tid < IN) {
        const int* mp = input_mask + (size_t)(o*IN + tid)*KBITS;
        unsigned pk =  (unsigned)(mp[0] & 127)
                    | ((unsigned)(mp[1] & 127) << 8)
                    | ((unsigned)(mp[2] & 127) << 16)
                    | ((unsigned)(mp[3] & 127) << 24);
        mks[tid] = pk;
    }

    const int b    = tid & 127;
    const int half = tid >> 7;
    // Per-thread sign bitmasks (128 bits per level) kept in VGPRs
    uint4 sp0 = ((const uint4*)spack)[b];
    uint4 sp1 = ((const uint4*)spack)[BATCH + b];
    __syncthreads();

    float acc = 0.0f;
#pragma unroll 4
    for (int ii = 0; ii < 64; ++ii) {
        int i = half*64 + ii;
        unsigned pk = mks[i];
        unsigned p0 = 0, p1 = 0;
#pragma unroll
        for (int k = 0; k < 4; ++k) {
            unsigned m  = (pk >> (8*k)) & 255u;
            unsigned w  = m >> 5;
            unsigned sh = m & 31u;
            unsigned w0 = (w < 2u) ? (w == 0u ? sp0.x : sp0.y)
                                   : (w == 2u ? sp0.z : sp0.w);
            unsigned w1 = (w < 2u) ? (w == 0u ? sp1.x : sp1.y)
                                   : (w == 2u ? sp1.z : sp1.w);
            p0 |= ((w0 >> sh) & 1u) << k;
            p1 |= ((w1 >> sh) & 1u) << k;
        }
        acc += gs[0][i][p0] + gs[1][i][p1];
    }
    partial[tid] = acc;
    __syncthreads();
    if (tid < BATCH) {
        out[(size_t)tid*OUT + o] = partial[tid] + partial[tid + 128] + bias[o];
    }
}

extern "C" void kernel_launch(void* const* d_in, const int* in_sizes, int n_in,
                              void* d_out, int out_size, void* d_ws, size_t ws_size,
                              hipStream_t stream) {
    (void)in_sizes; (void)n_in; (void)out_size; (void)ws_size;
    const float* x          = (const float*)d_in[0];
    const float* weight     = (const float*)d_in[1];
    const float* bias       = (const float*)d_in[2];
    const float* means      = (const float*)d_in[3];
    const int*   input_mask = (const int*)d_in[4];
    float* out = (float*)d_out;

    // workspace: [0,4096) sign bitmasks; [4096, 4096+2MB) transformed LUTs
    unsigned* spack = (unsigned*)d_ws;
    float*    g     = (float*)((char*)d_ws + 4096);

    signs_kernel<<<BATCH, IN, 0, stream>>>(x, means, spack);
    wmma_transform_kernel<<<T/16, 32, 0, stream>>>(weight, means, g);
    lut_gather_kernel<<<OUT, 256, 0, stream>>>(g, spack, input_mask, bias, out);
}
